// GCAModel_26800595927374
// MI455X (gfx1250) — matrix operand order repaired and checked by reading, not compiled, not run on hardware
//
#include <hip/hip_runtime.h>
#include <hip/hip_bf16.h>
#include <math.h>

typedef __bf16 bf16;
typedef __attribute__((ext_vector_type(16))) __bf16 v16bf;
typedef __attribute__((ext_vector_type(8)))  float  v8f;

#define B_   32
#define CDD_ 5
#define HIS_ 50
#define NIT_ 55      // CDD + HIS items per batch row
#define L_   32      // title length
#define E_   300     // embedding dim
#define EP_  320     // E padded to multiple of 32 (WMMA K)
#define H_   16      // heads
#define V_   16      // per-head value dim
#define R_   256     // repr dim = H*V
#define QD_  200     // keyProject out dim
#define QDP_ 208     // padded to multiple of 16
#define MASKED_HIS_ 40

static constexpr float kScale = 0.057735026918962574f; // 1/sqrt(300)

// ---- workspace layout (bytes) ----
static constexpr size_t WQT_OFF  = 0;                               // bf16 [16][320][320]
static constexpr size_t WQT_BYTES = (size_t)H_*EP_*EP_*2;           // 3,276,800
static constexpr size_t WVT_OFF  = WQT_OFF + WQT_BYTES;             // bf16 [256][320]
static constexpr size_t WVT_BYTES = (size_t)R_*EP_*2;               // 163,840
static constexpr size_t WKT_OFF  = WVT_OFF + WVT_BYTES;             // bf16 [208][256]
static constexpr size_t WKT_BYTES = (size_t)QDP_*R_*2;              // 106,496
static constexpr size_t REP_OFF  = WKT_OFF + WKT_BYTES;             // f32 [1760][256]
static constexpr size_t REP_BYTES = (size_t)B_*NIT_*R_*4;           // 1,802,240
static constexpr size_t HVAL_OFF = REP_OFF + REP_BYTES;             // f32 [32][50][32][256]
// total ~57.8 MB

// ---------------- WMMA fragment helpers (v_wmma_f32_16x16x32_bf16) -------------
// A 16x32 bf16: lane(0-15)=row M, kbase=0; lane(16-31)=row M, kbase=16.
// elems a[0..15] = K kbase..kbase+15.
__device__ inline v16bf ldA_bf(const bf16* base, int ld, int m0, int k0, int lane) {
  const bf16* p = base + (size_t)(m0 + (lane & 15)) * ld + k0 + ((lane >> 4) << 4);
  v16bf a;
#pragma unroll
  for (int j = 0; j < 16; ++j) a[j] = p[j];
  return a;
}
__device__ inline v16bf ldA_f32(const float* base, int ld, int m0, int k0, int lane) {
  const float* p = base + (size_t)(m0 + (lane & 15)) * ld + k0 + ((lane >> 4) << 4);
  v16bf a;
#pragma unroll
  for (int j = 0; j < 16; ++j) a[j] = (bf16)p[j];
  return a;
}
// B 32x16 bf16 from transposed storage T[n][k] (row stride ldk):
// lane(0-15)=col n, K 0..15 ; lane(16-31)=col n, K 16..31 -> contiguous 16 elems.
__device__ inline v16bf ldB_T(const bf16* baseT, int ldk, int n0, int k0, int lane) {
  const bf16* p = baseT + (size_t)(n0 + (lane & 15)) * ldk + k0 + ((lane >> 4) << 4);
  v16bf b;
#pragma unroll
  for (int j = 0; j < 16; ++j) b[j] = p[j];
  return b;
}
// B 32x16 bf16 from row-major storage S[k][n] (row stride ldn): strided per-lane reads.
__device__ inline v16bf ldB_rm(const bf16* base, int ldn, int n0, int k0, int lane) {
  const bf16* p = base + (size_t)(k0 + ((lane >> 4) << 4)) * ldn + n0 + (lane & 15);
  v16bf b;
#pragma unroll
  for (int j = 0; j < 16; ++j) b[j] = p[(size_t)j * ldn];
  return b;
}
__device__ inline v8f wmma_bf(v16bf a, v16bf b, v8f c) {
  return __builtin_amdgcn_wmma_f32_16x16x32_bf16(false, a, false, b, (short)0, c,
                                                 false, false);
}

// ---------------- kernel 0: weight transpose/pad to bf16 ----------------
__global__ __launch_bounds__(256) void prep_weights(
    const float* __restrict__ Wq, const float* __restrict__ Wv,
    const float* __restrict__ Wk, bf16* __restrict__ wqt,
    bf16* __restrict__ wvt, bf16* __restrict__ wkt) {
  int i = blockIdx.x * 256 + threadIdx.x;
  if (i < H_ * EP_ * EP_) {                 // wqt[h][f][e] = Wq[h][e][f]
    int h = i / (EP_ * EP_), rem = i % (EP_ * EP_);
    int f = rem / EP_, e = rem % EP_;
    float v = (f < E_ && e < E_) ? Wq[((size_t)h * E_ + e) * E_ + f] : 0.f;
    wqt[i] = (bf16)v;
  }
  if (i < R_ * EP_) {                       // wvt[h*16+v][e] = Wv[h][e][v]
    int nn = i / EP_, e = i % EP_;
    int h = nn >> 4, vv = nn & 15;
    float v = (e < E_) ? Wv[((size_t)h * E_ + e) * V_ + vv] : 0.f;
    wvt[i] = (bf16)v;
  }
  if (i < QDP_ * R_) {                      // wkt[qd][r] = Wk[r][qd]
    int qd = i / R_, r = i % R_;
    float v = (qd < QD_) ? Wk[(size_t)r * QD_ + qd] : 0.f;
    wkt[i] = (bf16)v;
  }
}

// ---------------- kernel 1: per-item encoder (1760 blocks, 8 wave32) ----------
__global__ __launch_bounds__(256) void encode_kernel(
    const int* __restrict__ cand_tok, const int* __restrict__ clk_tok,
    const float* __restrict__ emb, const bf16* __restrict__ wqt,
    const bf16* __restrict__ wvt, const bf16* __restrict__ wkt,
    const float* __restrict__ bk, const float* __restrict__ qw,
    float* __restrict__ rep_all, float* __restrict__ his_val) {
  __shared__ bf16  x_bf[L_ * EP_];    // 20.0 KB token embeddings (padded)
  __shared__ bf16  q_bf[L_ * EP_];    // 20.0 KB per-head query proj
  __shared__ bf16  xv_bf[L_ * R_];    // 16.0 KB XV = x @ Wv_concat
  __shared__ float s_f[L_ * L_];      //  4.0 KB attn scores / word weights
  __shared__ bf16  a_bf[L_ * L_];     //  2.0 KB softmaxed attn
  __shared__ float val_f[L_ * R_];    // 32.0 KB concat-head values
  __shared__ float wl[L_];            // word-attention logits (pre-scale)
  __shared__ float red[2];

  const int tid = threadIdx.x;
  const int wave = tid >> 5;
  const int lane = tid & 31;
  const int item = blockIdx.x;                 // b * 55 + n
  const int b = item / NIT_;
  const int n = item % NIT_;
  const int* tok = (n < CDD_) ? (cand_tok + ((size_t)b * CDD_ + n) * L_)
                              : (clk_tok + ((size_t)b * HIS_ + (n - CDD_)) * L_);

  // gather + bf16-convert token embeddings, zero-pad E->320
  for (int i = tid; i < L_ * EP_; i += 256) {
    int row = i / EP_, col = i % EP_;
    float v = (col < E_) ? emb[(size_t)tok[row] * E_ + col] : 0.f;
    x_bf[i] = (bf16)v;
  }
  __syncthreads();

  // XV = x @ Wv_concat : [32,320] x [320,256] -> [32,256]  (32 tiles)
  for (int tt = wave; tt < 2 * 16; tt += 8) {
    int mt = tt >> 4, nt = tt & 15;
    v8f acc = {};
#pragma unroll
    for (int kt = 0; kt < EP_ / 32; ++kt)
      acc = wmma_bf(ldA_bf(x_bf, EP_, mt * 16, kt * 32, lane),
                    ldB_T(wvt, EP_, nt * 16, kt * 32, lane), acc);
    int m0 = mt * 16 + ((lane >> 4) << 3), n0 = nt * 16 + (lane & 15);
#pragma unroll
    for (int j = 0; j < 8; ++j) xv_bf[(m0 + j) * R_ + n0] = (bf16)acc[j];
  }
  __syncthreads();

  // per-head attention
  for (int h = 0; h < H_; ++h) {
    // q = x @ Wq[h]^T-layout : [32,320] x [320,320] (40 tiles, the hot GEMM)
    const bf16* wq_h = wqt + (size_t)h * EP_ * EP_;
    for (int tt = wave; tt < 2 * 20; tt += 8) {
      int mt = tt / 20, nt = tt % 20;
      v8f acc = {};
#pragma unroll
      for (int kt = 0; kt < EP_ / 32; ++kt)
        acc = wmma_bf(ldA_bf(x_bf, EP_, mt * 16, kt * 32, lane),
                      ldB_T(wq_h, EP_, nt * 16, kt * 32, lane), acc);
      int m0 = mt * 16 + ((lane >> 4) << 3), n0 = nt * 16 + (lane & 15);
#pragma unroll
      for (int j = 0; j < 8; ++j) q_bf[(m0 + j) * EP_ + n0] = (bf16)acc[j];
    }
    __syncthreads();

    // s = (q @ x^T) * scale : [32,32], K=320 ; x row-major IS the transposed-B layout
    if (wave < 4) {
      int mt = wave >> 1, nt = wave & 1;
      v8f acc = {};
#pragma unroll
      for (int kt = 0; kt < EP_ / 32; ++kt)
        acc = wmma_bf(ldA_bf(q_bf, EP_, mt * 16, kt * 32, lane),
                      ldB_T(x_bf, EP_, nt * 16, kt * 32, lane), acc);
      int m0 = mt * 16 + ((lane >> 4) << 3), n0 = nt * 16 + (lane & 15);
#pragma unroll
      for (int j = 0; j < 8; ++j) s_f[(m0 + j) * L_ + n0] = acc[j] * kScale;
    }
    __syncthreads();

    // row softmax over 32 (one thread per row)
    if (tid < L_) {
      float mx = -1e30f;
      for (int m = 0; m < L_; ++m) mx = fmaxf(mx, s_f[tid * L_ + m]);
      float sum = 0.f;
      for (int m = 0; m < L_; ++m) sum += __expf(s_f[tid * L_ + m] - mx);
      float inv = 1.f / sum;
      for (int m = 0; m < L_; ++m)
        a_bf[tid * L_ + m] = (bf16)(__expf(s_f[tid * L_ + m] - mx) * inv);
    }
    __syncthreads();

    // v_h = a @ XV[:, h*16:(h+1)*16] : [32,32] x [32,16] (2 tiles, 1 WMMA each)
    if (wave < 2) {
      int mt = wave;
      v8f acc = {};
      acc = wmma_bf(ldA_bf(a_bf, L_, mt * 16, 0, lane),
                    ldB_rm(xv_bf, R_, h * 16, 0, lane), acc);
      int m0 = mt * 16 + ((lane >> 4) << 3), n0 = h * 16 + (lane & 15);
#pragma unroll
      for (int j = 0; j < 8; ++j) val_f[(m0 + j) * R_ + n0] = acc[j];
    }
    __syncthreads();
  }

  // word attention: wl[l] = sum_d qw[d] * tanh((val@Wk)[l,d] + bk[d])   (fused)
  if (tid < L_) wl[tid] = 0.f;
  __syncthreads();
  for (int tt = wave; tt < 2 * (QDP_ / 16); tt += 8) {   // 26 tiles, K=256
    int mt = tt / (QDP_ / 16), nt = tt % (QDP_ / 16);
    v8f acc = {};
#pragma unroll
    for (int kt = 0; kt < R_ / 32; ++kt)
      acc = wmma_bf(ldA_f32(val_f, R_, mt * 16, kt * 32, lane),
                    ldB_T(wkt, R_, nt * 16, kt * 32, lane), acc);
    int n0 = nt * 16 + (lane & 15);
    if (n0 < QD_) {
      float qwn = qw[n0], bkn = bk[n0];
      int m0 = mt * 16 + ((lane >> 4) << 3);
#pragma unroll
      for (int j = 0; j < 8; ++j)
        atomicAdd(&wl[m0 + j], qwn * tanhf(acc[j] + bkn));
    }
  }
  __syncthreads();

  // softmax over words, then rep[r] = sum_l ww[l] * val[l][r]
  if (tid == 0) {
    float mx = -1e30f;
    for (int l = 0; l < L_; ++l) mx = fmaxf(mx, wl[l] * kScale);
    float sum = 0.f;
    for (int l = 0; l < L_; ++l) sum += __expf(wl[l] * kScale - mx);
    red[0] = mx;
    red[1] = 1.f / sum;
  }
  __syncthreads();
  if (tid < L_) s_f[tid] = __expf(wl[tid] * kScale - red[0]) * red[1];
  __syncthreads();
  {
    float sum = 0.f;
    for (int l = 0; l < L_; ++l) sum += s_f[l] * val_f[l * R_ + tid];
    rep_all[(size_t)item * R_ + tid] = sum;
  }
  // persist his_val tiles (needed by the gather in kernel 2)
  if (n >= CDD_) {
    float4* dst = (float4*)(his_val + ((size_t)b * HIS_ + (n - CDD_)) * L_ * R_);
    const float4* src = (const float4*)val_f;
    for (int i = tid; i < L_ * R_ / 4; i += 256) dst[i] = src[i];
  }
}

// ---------- kernel 2: masked argmax of (cdd.his + gumbel), gather tile --------
__global__ __launch_bounds__(256) void select_kernel(
    const float* __restrict__ rep_all, const float* __restrict__ his_val,
    const float* __restrict__ gumbel, float* __restrict__ out) {
  __shared__ float cvec[R_];
  __shared__ float score[HIS_];
  __shared__ int best;
  const int bc = blockIdx.x;          // b*5 + c
  const int b = bc / CDD_, c = bc % CDD_;
  const int tid = threadIdx.x;
  cvec[tid] = rep_all[((size_t)b * NIT_ + c) * R_ + tid];
  __syncthreads();
  if (tid < HIS_) {
    float s;
    if (tid >= MASKED_HIS_) {
      s = -INFINITY;                  // masked_fill(-inf); gumbel can't revive it
    } else {
      const float* hv = rep_all + ((size_t)b * NIT_ + CDD_ + tid) * R_;
      float d = 0.f;
      for (int r = 0; r < R_; ++r) d += cvec[r] * hv[r];
      s = d + gumbel[(size_t)bc * HIS_ + tid];  // argmax invariant to /tau+softmax
    }
    score[tid] = s;
  }
  __syncthreads();
  if (tid == 0) {
    int bi = 0;
    float bs = score[0];
    for (int h = 1; h < HIS_; ++h)
      if (score[h] > bs) { bs = score[h]; bi = h; }
    best = bi;
  }
  __syncthreads();
  const float4* src = (const float4*)(his_val + ((size_t)b * HIS_ + best) * L_ * R_);
  float4* dst = (float4*)(out + (size_t)bc * L_ * R_);
  for (int i = tid; i < L_ * R_ / 4; i += 256) dst[i] = src[i];
}

extern "C" void kernel_launch(void* const* d_in, const int* in_sizes, int n_in,
                              void* d_out, int out_size, void* d_ws, size_t ws_size,
                              hipStream_t stream) {
  (void)in_sizes; (void)n_in; (void)out_size; (void)ws_size;
  const int*   cand_tok = (const int*)d_in[0];    // [32,5,32]
  const int*   clk_tok  = (const int*)d_in[1];    // [32,50,32]
  // d_in[2] his_mask, d_in[3..4] pads: unused by the reference math
  const float* gumbel   = (const float*)d_in[5];  // [32,5,50]
  const float* emb      = (const float*)d_in[6];  // [50000,300]
  const float* Wq       = (const float*)d_in[7];  // [16,300,300]
  const float* Wv       = (const float*)d_in[8];  // [16,300,16]
  const float* Wk       = (const float*)d_in[9];  // [256,200]
  const float* bk       = (const float*)d_in[10]; // [200]
  const float* qw       = (const float*)d_in[11]; // [1,200]
  float* out = (float*)d_out;                     // [32,5,32,256]

  char* ws = (char*)d_ws;
  bf16*  wqt     = (bf16*)(ws + WQT_OFF);
  bf16*  wvt     = (bf16*)(ws + WVT_OFF);
  bf16*  wkt     = (bf16*)(ws + WKT_OFF);
  float* rep_all = (float*)(ws + REP_OFF);
  float* his_val = (float*)(ws + HVAL_OFF);

  prep_weights<<<(H_ * EP_ * EP_ + 255) / 256, 256, 0, stream>>>(Wq, Wv, Wk, wqt,
                                                                 wvt, wkt);
  encode_kernel<<<B_ * NIT_, 256, 0, stream>>>(cand_tok, clk_tok, emb, wqt, wvt,
                                               wkt, bk, qw, rep_all, his_val);
  select_kernel<<<B_ * CDD_, 256, 0, stream>>>(rep_all, his_val, gumbel, out);
}
